// IMOR_40029095198936
// MI455X (gfx1250) — matrix operand order repaired and checked by reading, not compiled, not run on hardware
//
#include <hip/hip_runtime.h>

#define Bb 512
#define Ss 30
#define Cc 16
#define Hh 768
#define EPSF 1e-5f

typedef __attribute__((ext_vector_type(16))) __bf16 v16bf;
typedef __attribute__((ext_vector_type(8)))  float  v8f;
typedef __attribute__((address_space(3))) unsigned short lds_us;

union Frag {
  v16bf v;
  uint4 q[2];
};

__device__ __forceinline__ unsigned short f2bf(float f) {
  unsigned int u = __float_as_uint(f);
  u += 0x7FFFu + ((u >> 16) & 1u);
  return (unsigned short)(u >> 16);
}

// CDNA5 async copy: global -> LDS, 16B per lane, tracked by ASYNCcnt.
__device__ __forceinline__ void async_cp16(lds_us* dst, const unsigned short* src) {
  asm volatile("global_load_async_to_lds_b128 %0, %1, off"
               :
               : "v"(dst), "v"(src)
               : "memory");
}
__device__ __forceinline__ void async_wait0() {
  asm volatile("s_wait_asynccnt 0x0" ::: "memory");
}

// ---------------------------------------------------------------------------
// Weight transpose+convert:  dst[n*Ktot + koff + k] = bf16(src[k*768 + n])
// ---------------------------------------------------------------------------
__global__ void wconv(const float* __restrict__ src, unsigned short* __restrict__ dst,
                      int Ktot, int koff) {
  int i = blockIdx.x * 256 + threadIdx.x;
  if (i >= Hh * Hh) return;
  int k = i / Hh, n = i - k * Hh;
  dst[(size_t)n * Ktot + koff + k] = f2bf(src[(size_t)k * Hh + n]);
}

__global__ void f2both(const float* __restrict__ src, float* __restrict__ f,
                       unsigned short* __restrict__ bf, int n) {
  int i = blockIdx.x * 256 + threadIdx.x;
  if (i < n) { float v = src[i]; f[i] = v; bf[i] = f2bf(v); }
}

__global__ void refresh_bf(const float* __restrict__ f, unsigned short* __restrict__ bf, int n) {
  int i = blockIdx.x * 256 + threadIdx.x;
  if (i < n) bf[i] = f2bf(f[i]);
}

__global__ void extract_out(const float* __restrict__ cls_f, float* __restrict__ out) {
  int i = blockIdx.x * 256 + threadIdx.x;
  const int N = Bb * (Cc - 1) * Hh;
  if (i >= N) return;
  int h = i % Hh;
  int c = (i / Hh) % (Cc - 1) + 1;
  int b = i / (Hh * (Cc - 1));
  out[i] = cls_f[((size_t)b * Cc + c) * Hh + h];
}

// ---------------------------------------------------------------------------
// Normalized attention aggregates (cheap K-dims: S=30, C=16), fp32 math,
// bf16 outputs feeding the stacked-K WMMA GEMMs. Also emits bias coefficients
// c = rowsum/(rowsum+eps) so the relation biases fold into the GEMM epilogue.
// ---------------------------------------------------------------------------
__global__ __launch_bounds__(256) void aggregate(
    const float* __restrict__ slots_f, const float* __restrict__ cls_f,
    const float* __restrict__ a_cur, const float* __restrict__ a_slot,
    const float* __restrict__ a_last, const float* __restrict__ a_dom,
    unsigned short* __restrict__ agg_s,   // 4 matrices of [B*S,H], r-major
    unsigned short* __restrict__ agg_c0, unsigned short* __restrict__ agg_c2,
    float* __restrict__ coeff_s, float* __restrict__ coeff_c)
{
  const int b = blockIdx.x;
  const int t = threadIdx.x;
  __shared__ float ac[Ss * Cc], al[Ss * Cc];
  __shared__ float asl[Ss * Ss], ad[Ss * Ss];
  __shared__ float inv_rs[4][Ss];
  __shared__ float inv_cs[2][Cc];

  for (int i = t; i < Ss * Cc; i += 256) {
    ac[i] = a_cur[(size_t)b * Ss * Cc + i];
    al[i] = a_last[(size_t)b * Ss * Cc + i];
  }
  for (int i = t; i < Ss * Ss; i += 256) {
    asl[i] = a_slot[(size_t)b * Ss * Ss + i];
    ad[i]  = a_dom[(size_t)b * Ss * Ss + i];
  }
  __syncthreads();
  if (t < Ss) {
    float s0 = 0.f, s1 = 0.f, s2 = 0.f, s3 = 0.f;
    for (int c = 0; c < Cc; ++c) { s0 += ac[t * Cc + c]; s2 += al[t * Cc + c]; }
    for (int j = 0; j < Ss; ++j) { s1 += asl[t * Ss + j]; s3 += ad[t * Ss + j]; }
    inv_rs[0][t] = 1.f / (s0 + EPSF);
    inv_rs[1][t] = 1.f / (s1 + EPSF);
    inv_rs[2][t] = 1.f / (s2 + EPSF);
    inv_rs[3][t] = 1.f / (s3 + EPSF);
    float* cs = coeff_s + ((size_t)b * Ss + t) * 4;
    cs[0] = s0 / (s0 + EPSF); cs[1] = s1 / (s1 + EPSF);
    cs[2] = s2 / (s2 + EPSF); cs[3] = s3 / (s3 + EPSF);
  } else if (t >= 32 && t < 32 + Cc) {
    const int c = t - 32;
    float s0 = 0.f, s2 = 0.f;
    for (int s = 0; s < Ss; ++s) { s0 += ac[s * Cc + c]; s2 += al[s * Cc + c]; }
    inv_cs[0][c] = 1.f / (s0 + EPSF);
    inv_cs[1][c] = 1.f / (s2 + EPSF);
    float* cc = coeff_c + ((size_t)b * Cc + c) * 2;
    cc[0] = s0 / (s0 + EPSF); cc[1] = s2 / (s2 + EPSF);
  }
  __syncthreads();

  const size_t SMH = (size_t)Bb * Ss * Hh;
  for (int hb = 0; hb < Hh / 256; ++hb) {
    const int h = t + hb * 256;
    float cl[Cc], sl[Ss];
#pragma unroll
    for (int c = 0; c < Cc; ++c) cl[c] = cls_f[((size_t)b * Cc + c) * Hh + h];
#pragma unroll
    for (int s = 0; s < Ss; ++s) sl[s] = slots_f[((size_t)b * Ss + s) * Hh + h];
    for (int s = 0; s < Ss; ++s) {
      float v0 = 0.f, v1 = 0.f, v2 = 0.f, v3 = 0.f;
#pragma unroll
      for (int c = 0; c < Cc; ++c) { v0 += ac[s * Cc + c] * cl[c]; v2 += al[s * Cc + c] * cl[c]; }
#pragma unroll
      for (int j = 0; j < Ss; ++j) { v1 += asl[s * Ss + j] * sl[j]; v3 += ad[s * Ss + j] * sl[j]; }
      const size_t o = ((size_t)b * Ss + s) * Hh + h;
      agg_s[0 * SMH + o] = f2bf(v0 * inv_rs[0][s]);
      agg_s[1 * SMH + o] = f2bf(v1 * inv_rs[1][s]);
      agg_s[2 * SMH + o] = f2bf(v2 * inv_rs[2][s]);
      agg_s[3 * SMH + o] = f2bf(v3 * inv_rs[3][s]);
    }
    for (int c = 0; c < Cc; ++c) {
      float v0 = 0.f, v2 = 0.f;
#pragma unroll
      for (int s = 0; s < Ss; ++s) { v0 += ac[s * Cc + c] * sl[s]; v2 += al[s * Cc + c] * sl[s]; }
      const size_t o = ((size_t)b * Cc + c) * Hh + h;
      agg_c0[o] = f2bf(v0 * inv_cs[0][c]);
      agg_c2[o] = f2bf(v2 * inv_cs[1][c]);
    }
  }
}

// ---------------------------------------------------------------------------
// Stacked-K bf16 WMMA GEMM.  D[M,768] = sum_blk A_blk[M,768] @ Wt_blk
// Wt is pre-transposed [768(n), Ktot(k)] bf16.  Block tile 128x128, 8 waves,
// each wave 4x2 WMMA accumulators, K-step 32.  A tiles are double-buffered in
// LDS via CDNA5 async copies (global_load_async_to_lds_b128 / ASYNCcnt):
//   issue async into back buffer -> WMMA on front buffer
//   -> s_wait_asynccnt 0 -> barrier -> flip.
// mode 0: slots projection epilogue  (+b_s + sum_r coeff[r]*b_r[r])
// mode 1: cls projection epilogue    (+b_s + c0*b_r[0] + c1*b_r[2])
// mode 2: gate epilogue              (sigmoid gate, relu residual, in-place X)
// ---------------------------------------------------------------------------
__global__ __launch_bounds__(256) void gemm_wmma_bf16(
    const unsigned short* __restrict__ A0, const unsigned short* __restrict__ A1,
    const unsigned short* __restrict__ A2, const unsigned short* __restrict__ A3,
    const unsigned short* __restrict__ A4,
    int nblk, const unsigned short* __restrict__ Wt,
    const float* __restrict__ bias, const float* __restrict__ brel,
    const float* __restrict__ coeff,
    float* __restrict__ outF, unsigned short* __restrict__ outBf,
    const float* __restrict__ Uf, float* __restrict__ Xf, int mode)
{
  constexpr int LDA = 40;                         // 32 + 8 pad (bf16 elems)
  __shared__ alignas(16) unsigned short As[2][128 * LDA];

  const unsigned short* Ablk[5] = {A0, A1, A2, A3, A4};
  const int tid  = threadIdx.x;
  const int lane = tid & 31;
  const int wave = tid >> 5;
  const int half = lane >> 4;                     // lane group (K-interleave)
  const int lr   = lane & 15;
  const int wm   = wave >> 2;                     // 0..1 -> 64-row group
  const int wn   = wave & 3;                      // 0..3 -> 32-col group
  const int m0   = blockIdx.x * 128;
  const int n0   = blockIdx.y * 128;
  const int Ktot = nblk * Hh;

  // Hint the L2-resident weight panel into cache (global_prefetch_b8).
  __builtin_prefetch(Wt + (size_t)(n0 + wn * 32 + lr) * Ktot, 0, 1);

  v8f acc[4][2];
#pragma unroll
  for (int i = 0; i < 4; ++i)
#pragma unroll
    for (int j = 0; j < 2; ++j)
#pragma unroll
      for (int e = 0; e < 8; ++e) acc[i][j][e] = 0.f;

  // A staging: 128 rows x 32 cols bf16 = 512 x 16B chunks; 2 per thread
  const int r0 = tid >> 2,         o0 = (tid & 3) * 8;
  const int r1 = (tid + 256) >> 2, o1 = ((tid + 256) & 3) * 8;

  const int nsteps = nblk * (Hh / 32);

  { // stage step 0 (blk 0, kb 0) via async copy
    const unsigned short* Ab = Ablk[0];
    async_cp16((lds_us*)&As[0][r0 * LDA + o0], Ab + (size_t)(m0 + r0) * Hh + o0);
    async_cp16((lds_us*)&As[0][r1 * LDA + o1], Ab + (size_t)(m0 + r1) * Hh + o1);
  }
  async_wait0();
  __syncthreads();

  for (int s = 0; s < nsteps; ++s) {
    const int buf = s & 1;
    if (s + 1 < nsteps) {                         // async-prefetch next A tile
      const int sn   = s + 1;
      const int blkN = sn / 24;
      const int kbN  = (sn % 24) * 32;
      const unsigned short* Ab = Ablk[blkN];
      async_cp16((lds_us*)&As[buf ^ 1][r0 * LDA + o0],
                 Ab + (size_t)(m0 + r0) * Hh + kbN + o0);
      async_cp16((lds_us*)&As[buf ^ 1][r1 * LDA + o1],
                 Ab + (size_t)(m0 + r1) * Hh + kbN + o1);
    }
    const int blk = s / 24;
    const int kb  = (s % 24) * 32;

    // B fragments straight from L2-resident transposed weights:
    // lane lr -> column n, lane<16 holds K {0..7,16..23}, lane>=16 holds {8..15,24..31}.
    Frag bf0, bf1;
    {
      const unsigned short* pb = Wt + (size_t)(n0 + wn * 32 + lr) * Ktot + blk * Hh + kb + half * 8;
      bf0.q[0] = *(const uint4*)(pb);
      bf0.q[1] = *(const uint4*)(pb + 16);
      const unsigned short* pb1 = pb + (size_t)16 * Ktot;
      bf1.q[0] = *(const uint4*)(pb1);
      bf1.q[1] = *(const uint4*)(pb1 + 16);
    }
#pragma unroll
    for (int mi = 0; mi < 4; ++mi) {
      // A fragment: lane<16 holds K {0..7,16..23}, lane>=16 holds {8..15,24..31}
      Frag af;
      const unsigned short* pa = &As[buf][(wm * 64 + mi * 16 + lr) * LDA + half * 8];
      af.q[0] = *(const uint4*)(pa);
      af.q[1] = *(const uint4*)(pa + 16);
      acc[mi][0] = __builtin_amdgcn_wmma_f32_16x16x32_bf16(
          false, af.v, false, bf0.v, (short)0, acc[mi][0], false, false);
      acc[mi][1] = __builtin_amdgcn_wmma_f32_16x16x32_bf16(
          false, af.v, false, bf1.v, (short)0, acc[mi][1], false, false);
    }
    async_wait0();                                // publish back buffer
    __syncthreads();
  }

  // Epilogue: D layout -> VGPR j holds row (half*8 + j), lane lr = column.
#pragma unroll
  for (int mi = 0; mi < 4; ++mi) {
#pragma unroll
    for (int ni = 0; ni < 2; ++ni) {
      const int col = n0 + wn * 32 + ni * 16 + lr;
#pragma unroll
      for (int j = 0; j < 8; ++j) {
        const int row = m0 + wm * 64 + mi * 16 + half * 8 + j;
        const size_t o = (size_t)row * Hh + col;
        float v = acc[mi][ni][j];
        if (mode == 0) {
          v += bias[col];
          v += coeff[row * 4 + 0] * brel[0 * Hh + col];
          v += coeff[row * 4 + 1] * brel[1 * Hh + col];
          v += coeff[row * 4 + 2] * brel[2 * Hh + col];
          v += coeff[row * 4 + 3] * brel[3 * Hh + col];
          outF[o]  = v;
          outBf[o] = f2bf(v);
        } else if (mode == 1) {
          v += bias[col];
          v += coeff[row * 2 + 0] * brel[0 * Hh + col];
          v += coeff[row * 2 + 1] * brel[2 * Hh + col];
          outF[o]  = v;
          outBf[o] = f2bf(v);
        } else {
          v += bias[col];
          const float g  = 1.f / (1.f + __expf(-v));
          const float xu = Uf[o];
          const float x  = Xf[o];
          Xf[o] = fmaxf(xu, 0.f) * g + x * (1.f - g);
        }
      }
    }
  }
}

// ---------------------------------------------------------------------------
extern "C" void kernel_launch(void* const* d_in, const int* in_sizes, int n_in,
                              void* d_out, int out_size, void* d_ws, size_t ws_size,
                              hipStream_t stream) {
  const float* slots  = (const float*)d_in[0];
  const float* cls    = (const float*)d_in[1];
  const float* a_cur  = (const float*)d_in[2];
  const float* a_slot = (const float*)d_in[3];
  const float* a_last = (const float*)d_in[4];
  const float* a_dom  = (const float*)d_in[5];
  const float* W_r    = (const float*)d_in[6];
  const float* b_r    = (const float*)d_in[7];
  const float* W_s    = (const float*)d_in[8];
  const float* b_s    = (const float*)d_in[9];
  const float* W_g    = (const float*)d_in[10];
  const float* b_g    = (const float*)d_in[11];
  float* out = (float*)d_out;

  const size_t SMH = (size_t)Bb * Ss * Hh;   // 11,796,480
  const size_t CMH = (size_t)Bb * Cc * Hh;   //  6,291,456
  const int HH = Hh * Hh;                    //    589,824

  char* p = (char*)d_ws;
  auto alloc = [&](size_t bytes) -> void* {
    void* r = (void*)p;
    p += (bytes + 255) & ~(size_t)255;
    return r;
  };
  float* slots_f = (float*)alloc(SMH * 4);
  float* cls_f   = (float*)alloc(CMH * 4);
  float* slots_u = (float*)alloc(SMH * 4);
  float* cls_u   = (float*)alloc(CMH * 4);
  unsigned short* slots_bf   = (unsigned short*)alloc(SMH * 2);
  unsigned short* cls_bf     = (unsigned short*)alloc(CMH * 2);
  unsigned short* slots_u_bf = (unsigned short*)alloc(SMH * 2);
  unsigned short* cls_u_bf   = (unsigned short*)alloc(CMH * 2);
  unsigned short* agg_s      = (unsigned short*)alloc(4 * SMH * 2);
  unsigned short* agg_c0     = (unsigned short*)alloc(CMH * 2);
  unsigned short* agg_c2     = (unsigned short*)alloc(CMH * 2);
  float* coeff_s = (float*)alloc((size_t)Bb * Ss * 4 * 4);
  float* coeff_c = (float*)alloc((size_t)Bb * Cc * 2 * 4);
  unsigned short* Wt_s = (unsigned short*)alloc((size_t)5 * HH * 2);
  unsigned short* Wt_c = (unsigned short*)alloc((size_t)3 * HH * 2);
  unsigned short* Wt_g = (unsigned short*)alloc((size_t)2 * HH * 2);

  dim3 cgrid((HH + 255) / 256), cblk(256);
  // slots stack [W_s; W_r0..3], Ktot = 5H
  wconv<<<cgrid, cblk, 0, stream>>>(W_s, Wt_s, 5 * Hh, 0);
  for (int r = 0; r < 4; ++r)
    wconv<<<cgrid, cblk, 0, stream>>>(W_r + (size_t)r * HH, Wt_s, 5 * Hh, (r + 1) * Hh);
  // cls stack [W_s; W_r0; W_r2], Ktot = 3H
  wconv<<<cgrid, cblk, 0, stream>>>(W_s, Wt_c, 3 * Hh, 0);
  wconv<<<cgrid, cblk, 0, stream>>>(W_r, Wt_c, 3 * Hh, Hh);
  wconv<<<cgrid, cblk, 0, stream>>>(W_r + (size_t)2 * HH, Wt_c, 3 * Hh, 2 * Hh);
  // gate stack [W_g rows 0..H-1 ; rows H..2H-1] as two K-blocks, Ktot = 2H
  wconv<<<cgrid, cblk, 0, stream>>>(W_g, Wt_g, 2 * Hh, 0);
  wconv<<<cgrid, cblk, 0, stream>>>(W_g + (size_t)HH, Wt_g, 2 * Hh, Hh);

  f2both<<<(int)((SMH + 255) / 256), 256, 0, stream>>>(slots, slots_f, slots_bf, (int)SMH);
  f2both<<<(int)((CMH + 255) / 256), 256, 0, stream>>>(cls, cls_f, cls_bf, (int)CMH);

  for (int layer = 0; layer < 2; ++layer) {
    aggregate<<<Bb, 256, 0, stream>>>(slots_f, cls_f, a_cur, a_slot, a_last, a_dom,
                                      agg_s, agg_c0, agg_c2, coeff_s, coeff_c);
    // slots projection: M = B*S = 15360
    gemm_wmma_bf16<<<dim3(15360 / 128, Hh / 128), 256, 0, stream>>>(
        slots_bf, agg_s, agg_s + SMH, agg_s + 2 * SMH, agg_s + 3 * SMH,
        5, Wt_s, b_s, b_r, coeff_s, slots_u, slots_u_bf, nullptr, nullptr, 0);
    // cls projection: M = B*C = 8192
    gemm_wmma_bf16<<<dim3(8192 / 128, Hh / 128), 256, 0, stream>>>(
        cls_bf, agg_c0, agg_c2, agg_c2, agg_c2,
        3, Wt_c, b_s, b_r, coeff_c, cls_u, cls_u_bf, nullptr, nullptr, 1);
    // gates (in-place fp32 state update)
    gemm_wmma_bf16<<<dim3(15360 / 128, Hh / 128), 256, 0, stream>>>(
        slots_u_bf, slots_bf, slots_bf, slots_bf, slots_bf,
        2, Wt_g, b_g, nullptr, nullptr, nullptr, nullptr, slots_u, slots_f, 2);
    gemm_wmma_bf16<<<dim3(8192 / 128, Hh / 128), 256, 0, stream>>>(
        cls_u_bf, cls_bf, cls_bf, cls_bf, cls_bf,
        2, Wt_g, b_g, nullptr, nullptr, nullptr, nullptr, cls_u, cls_f, 2);
    if (layer == 0) {  // refresh bf16 state copies for the next layer
      refresh_bf<<<(int)((SMH + 255) / 256), 256, 0, stream>>>(slots_f, slots_bf, (int)SMH);
      refresh_bf<<<(int)((CMH + 255) / 256), 256, 0, stream>>>(cls_f, cls_bf, (int)CMH);
    }
  }
  const int NOUT = Bb * (Cc - 1) * Hh;
  extract_out<<<(NOUT + 255) / 256, 256, 0, stream>>>(cls_f, out);
}
